// PPO_81183471829356
// MI455X (gfx1250) — matrix operand order repaired and checked by reading, not compiled, not run on hardware
//
#include <hip/hip_runtime.h>
#include <hip/hip_bf16.h>
#include <math.h>

#define T_STEPS 65536
#define D_IN    512
#define H_DIM   256
#define N_ACT   16
#define CS      256
#define NCH     (T_STEPS / CS)
#define GAMMA_F 0.99f
#define LAM_F   0.95f
#define CLIP_F  0.2f

// swizzled weight sizes (bf16 elements)
#define W1S_ELEMS (16 * 16 * 32 * 16)  // [kt=16][nt=16][lane=32][e=16]
#define W2S_ELEMS (8 * 16 * 32 * 16)   // [kt=8 ][nt=16][lane=32][e=16]
#define WHS_ELEMS (8 * 2 * 32 * 16)    // [kt=8 ][nt=2 ][lane=32][e=16]  (Wa | Wc,0..)

typedef __attribute__((ext_vector_type(16))) __bf16 v16bf;
typedef __attribute__((ext_vector_type(8)))  __bf16 v8bf;
typedef __attribute__((ext_vector_type(8)))  float  v8f;

static __device__ __forceinline__ v8f wmma_bf16(v16bf a, v16bf b, v8f c) {
  return __builtin_amdgcn_wmma_f32_16x16x32_bf16(false, a, false, b, (short)0, c,
                                                 false, false);
}
static __device__ __forceinline__ v16bf cat8(v8bf lo, v8bf hi) {
  return __builtin_shufflevector(lo, hi, 0, 1, 2, 3, 4, 5, 6, 7,
                                 8, 9, 10, 11, 12, 13, 14, 15);
}

// ---------------------------------------------------------------------------
// ws layout (floats unless noted):
//  [0..7] accumulators: adv_sum, adv_sumsq, pg, ent, v, mean, inv_std, pad
//  [8 .. 8+T)  adv ; then P[NCH], Q[NCH], C[NCH]
//  then bf16: W1s, W2s, Whs (16B-aligned)
// ---------------------------------------------------------------------------

__global__ void k_zero(float* acc) {
  if (threadIdx.x < 8) acc[threadIdx.x] = 0.f;
}

// Convert weights to bf16 in WMMA B-fragment order: chunk index
// ((kt*NT + nt)*32 + lane)*16 + e  holds  W[(kt*32 + (lane>>4)*16 + e)][nt*16 + (lane&15)]
__global__ void k_prep(const float* __restrict__ W1, const float* __restrict__ W2,
                       const float* __restrict__ Wa, const float* __restrict__ Wc,
                       __bf16* __restrict__ W1s, __bf16* __restrict__ W2s,
                       __bf16* __restrict__ Whs) {
  int i = blockIdx.x * blockDim.x + threadIdx.x;
  if (i < W1S_ELEMS) {
    int e = i & 15, lane = (i >> 4) & 31, nt = (i >> 9) & 15, kt = i >> 13;
    int k = kt * 32 + (lane >> 4) * 16 + e;
    int c = nt * 16 + (lane & 15);
    W1s[i] = (__bf16)W1[(size_t)k * H_DIM + c];
  }
  if (i < W2S_ELEMS) {
    int e = i & 15, lane = (i >> 4) & 31, nt = (i >> 9) & 15, kt = i >> 13;
    int k = kt * 32 + (lane >> 4) * 16 + e;
    int c = nt * 16 + (lane & 15);
    W2s[i] = (__bf16)W2[(size_t)k * H_DIM + c];
  }
  if (i < WHS_ELEMS) {
    int e = i & 15, lane = (i >> 4) & 31, nt = (i >> 9) & 1, kt = i >> 10;
    int k = kt * 32 + (lane >> 4) * 16 + e;
    int n = lane & 15;
    float v = 0.f;
    if (nt == 0) v = Wa[(size_t)k * N_ACT + n];       // actor columns
    else if (n == 0) v = Wc[k];                        // critic as column 0
    Whs[i] = (__bf16)v;
  }
}

// ---- GAE: chunked affine-composition scan -------------------------------
__global__ void k_gae_compose(const float* __restrict__ rewards,
                              const float* __restrict__ terms,
                              const float* __restrict__ values,
                              float* __restrict__ P, float* __restrict__ Q) {
  int ch = blockIdx.x * blockDim.x + threadIdx.x;
  if (ch >= NCH) return;
  int s = ch * CS, e = s + CS;
  float p = 1.f, q = 0.f;
  for (int t = e - 1; t >= s; --t) {
    float nt = 1.f - terms[t];
    float nv = (t + 1 < T_STEPS) ? values[t + 1] : 0.f;
    float delta = rewards[t] + GAMMA_F * nv * nt - values[t];
    float c = GAMMA_F * LAM_F * nt;
    q = delta + c * q;
    p = c * p;
  }
  P[ch] = p;
  Q[ch] = q;
}

__global__ void k_gae_carry(const float* __restrict__ P,
                            const float* __restrict__ Q,
                            float* __restrict__ carry_in) {
  if (threadIdx.x != 0 || blockIdx.x != 0) return;
  float carry = 0.f;
  for (int ch = NCH - 1; ch >= 0; --ch) {
    carry_in[ch] = carry;
    carry = Q[ch] + P[ch] * carry;
  }
}

__global__ void k_gae_apply(const float* __restrict__ rewards,
                            const float* __restrict__ terms,
                            const float* __restrict__ values,
                            const float* __restrict__ carry_in,
                            float* __restrict__ adv, float* __restrict__ acc) {
  int ch = blockIdx.x * blockDim.x + threadIdx.x;
  if (ch >= NCH) return;
  int s = ch * CS, e = s + CS;
  float x = carry_in[ch];
  float sum = 0.f, sumsq = 0.f;
  for (int t = e - 1; t >= s; --t) {
    float nt = 1.f - terms[t];
    float nv = (t + 1 < T_STEPS) ? values[t + 1] : 0.f;
    float delta = rewards[t] + GAMMA_F * nv * nt - values[t];
    x = delta + GAMMA_F * LAM_F * nt * x;
    adv[t] = x;
    sum += x;
    sumsq += x * x;
  }
  atomicAdd(&acc[0], sum);
  atomicAdd(&acc[1], sumsq);
}

__global__ void k_gae_stats(float* acc) {
  if (threadIdx.x != 0 || blockIdx.x != 0) return;
  float mean = acc[0] / (float)T_STEPS;
  float var  = acc[1] / (float)T_STEPS - mean * mean;
  float sd   = sqrtf(fmaxf(var, 0.f));
  acc[5] = mean;
  acc[6] = 1.f / (sd + 1e-8f);
}

// ---------------------------------------------------------------------------
// Fused MLP + PPO loss: 64-row tiles, 128 threads (4 waves x 16 rows).
// ---------------------------------------------------------------------------
__global__ __launch_bounds__(128) void k_ppo_main(
    const float* __restrict__ obs, const int* __restrict__ actions,
    const float* __restrict__ logprobs, const float* __restrict__ values,
    const __bf16* __restrict__ W1s, const float* __restrict__ b1,
    const __bf16* __restrict__ W2s, const float* __restrict__ b2,
    const __bf16* __restrict__ Whs, const float* __restrict__ ba,
    const float* __restrict__ bc,
    const float* __restrict__ adv, float* __restrict__ acc) {
  __shared__ __bf16 sh_x[64][56];           // obs K-slice, stride 112B (16B aligned)
  __shared__ __bf16 sh_wB[16 * 32 * 16];    // one pre-swizzled weight K-slice (16KB)
  __shared__ __bf16 sh_h[64][H_DIM + 8];    // hidden, stride 528B (16B aligned)
  __shared__ float  sh_logits[64][N_ACT + 1];
  __shared__ float  sh_val[64];
  __shared__ float  sh_red[128];

  const int tid  = threadIdx.x;
  const int wave = tid >> 5;
  const int lane = tid & 31;
  const int row0 = blockIdx.x * 64;
  const int m    = lane & 15;
  const int half = lane >> 4;
  const int lr   = wave * 16 + m;

  const v8f zero8 = {0.f, 0.f, 0.f, 0.f, 0.f, 0.f, 0.f, 0.f};

  // ---------------- layer 1: h1 = tanh(obs @ W1 + b1) ----------------
  v8f acc1[16];
#pragma unroll
  for (int i = 0; i < 16; ++i) acc1[i] = zero8;

  for (int kt = 0; kt < D_IN / 32; ++kt) {
    const int kb = kt * 32;
    __syncthreads();
    {  // stage obs slice: float4 loads -> bf16 -> v8bf stores
      const int r = tid >> 1, c0 = (tid & 1) * 16;
      const float4* gp = (const float4*)(obs + (size_t)(row0 + r) * D_IN + kb + c0);
      float4 f0 = gp[0], f1 = gp[1], f2 = gp[2], f3 = gp[3];
      v8bf o0, o1;
      o0[0] = (__bf16)f0.x; o0[1] = (__bf16)f0.y; o0[2] = (__bf16)f0.z; o0[3] = (__bf16)f0.w;
      o0[4] = (__bf16)f1.x; o0[5] = (__bf16)f1.y; o0[6] = (__bf16)f1.z; o0[7] = (__bf16)f1.w;
      o1[0] = (__bf16)f2.x; o1[1] = (__bf16)f2.y; o1[2] = (__bf16)f2.z; o1[3] = (__bf16)f2.w;
      o1[4] = (__bf16)f3.x; o1[5] = (__bf16)f3.y; o1[6] = (__bf16)f3.z; o1[7] = (__bf16)f3.w;
      *(v8bf*)&sh_x[r][c0]     = o0;
      *(v8bf*)&sh_x[r][c0 + 8] = o1;
    }
    {  // stage pre-swizzled weight slice: straight uint4 copy (16KB)
      const uint4* src = (const uint4*)(W1s + kt * (16 * 32 * 16));
      uint4* dst = (uint4*)sh_wB;
#pragma unroll
      for (int j = 0; j < 8; ++j) dst[tid + 128 * j] = src[tid + 128 * j];
    }
    if (kt + 1 < D_IN / 32)
      __builtin_prefetch(W1s + (kt + 1) * (16 * 32 * 16), 0, 1);
    __syncthreads();

    v16bf a = cat8(*(const v8bf*)&sh_x[lr][half * 16],
                   *(const v8bf*)&sh_x[lr][half * 16 + 8]);
#pragma unroll
    for (int nt = 0; nt < 16; ++nt) {
      v16bf b = *(const v16bf*)&sh_wB[(nt * 32 + lane) * 16];
      acc1[nt] = wmma_bf16(a, b, acc1[nt]);
    }
  }
  __syncthreads();
#pragma unroll
  for (int nt = 0; nt < 16; ++nt) {
    const int col = nt * 16 + m;
    const float bias = b1[col];
#pragma unroll
    for (int r = 0; r < 8; ++r)
      sh_h[wave * 16 + half * 8 + r][col] = (__bf16)tanhf(acc1[nt][r] + bias);
  }

  // ---------------- layer 2: h2 = tanh(h1 @ W2 + b2) ----------------
  v8f acc2[16];
#pragma unroll
  for (int i = 0; i < 16; ++i) acc2[i] = zero8;

  for (int kt = 0; kt < H_DIM / 32; ++kt) {
    const int kb = kt * 32;
    __syncthreads();
    {
      const uint4* src = (const uint4*)(W2s + kt * (16 * 32 * 16));
      uint4* dst = (uint4*)sh_wB;
#pragma unroll
      for (int j = 0; j < 8; ++j) dst[tid + 128 * j] = src[tid + 128 * j];
    }
    __syncthreads();

    v16bf a = cat8(*(const v8bf*)&sh_h[lr][kb + half * 16],
                   *(const v8bf*)&sh_h[lr][kb + half * 16 + 8]);
#pragma unroll
    for (int nt = 0; nt < 16; ++nt) {
      v16bf b = *(const v16bf*)&sh_wB[(nt * 32 + lane) * 16];
      acc2[nt] = wmma_bf16(a, b, acc2[nt]);
    }
  }
  __syncthreads();
#pragma unroll
  for (int nt = 0; nt < 16; ++nt) {
    const int col = nt * 16 + m;
    const float bias = b2[col];
#pragma unroll
    for (int r = 0; r < 8; ++r)
      sh_h[wave * 16 + half * 8 + r][col] = (__bf16)tanhf(acc2[nt][r] + bias);
  }

  // -------- heads: [logits | value] = h2 @ [Wa | Wc,0..] + bias --------
  v8f acch[2] = {zero8, zero8};
  for (int kt = 0; kt < H_DIM / 32; ++kt) {
    const int kb = kt * 32;
    v16bf a = cat8(*(const v8bf*)&sh_h[lr][kb + half * 16],
                   *(const v8bf*)&sh_h[lr][kb + half * 16 + 8]);
#pragma unroll
    for (int nt = 0; nt < 2; ++nt) {
      v16bf b = *(const v16bf*)&Whs[((kt * 2 + nt) * 32 + lane) * 16];
      acch[nt] = wmma_bf16(a, b, acch[nt]);
    }
  }
  {
    const float bias = ba[m];
#pragma unroll
    for (int r = 0; r < 8; ++r)
      sh_logits[wave * 16 + half * 8 + r][m] = acch[0][r] + bias;
    if (m == 0) {
#pragma unroll
      for (int r = 0; r < 8; ++r)
        sh_val[wave * 16 + half * 8 + r] = acch[1][r] + bc[0];
    }
  }
  __syncthreads();

  // ---------------- per-row PPO loss terms ----------------
  float pg = 0.f, ent = 0.f, vl = 0.f;
  if (tid < 64) {
    const int r = tid;
    const int t = row0 + r;
    const float v = sh_val[r];
    float l[N_ACT];
    float mx = -3.0e38f;
#pragma unroll
    for (int i = 0; i < N_ACT; ++i) { l[i] = sh_logits[r][i]; mx = fmaxf(mx, l[i]); }
    float se = 0.f;
#pragma unroll
    for (int i = 0; i < N_ACT; ++i) se += expf(l[i] - mx);
    const float lse = mx + logf(se);
#pragma unroll
    for (int i = 0; i < N_ACT; ++i) {
      const float lp = l[i] - lse;
      ent -= expf(lp) * lp;
    }
    const float newlp = l[actions[t]] - lse;
    const float advv  = adv[t];
    const float an    = (advv - acc[5]) * acc[6];
    const float ratio = expf(newlp - logprobs[t]);
    const float rcl   = fminf(fmaxf(ratio, 1.f - CLIP_F), 1.f + CLIP_F);
    pg = fmaxf(-an * ratio, -an * rcl);
    const float vo  = values[t];
    const float ret = advv + vo;
    const float d1  = v - ret;
    const float d2  = vo + fminf(fmaxf(v - vo, -CLIP_F), CLIP_F) - ret;
    vl = fmaxf(d1 * d1, d2 * d2);
  }

  __syncthreads();
  sh_red[tid] = pg;
  __syncthreads();
  for (int s = 64; s >= 1; s >>= 1) {
    if (tid < s) sh_red[tid] += sh_red[tid + s];
    __syncthreads();
  }
  if (tid == 0) atomicAdd(&acc[2], sh_red[0]);
  __syncthreads();
  sh_red[tid] = ent;
  __syncthreads();
  for (int s = 64; s >= 1; s >>= 1) {
    if (tid < s) sh_red[tid] += sh_red[tid + s];
    __syncthreads();
  }
  if (tid == 0) atomicAdd(&acc[3], sh_red[0]);
  __syncthreads();
  sh_red[tid] = vl;
  __syncthreads();
  for (int s = 64; s >= 1; s >>= 1) {
    if (tid < s) sh_red[tid] += sh_red[tid + s];
    __syncthreads();
  }
  if (tid == 0) atomicAdd(&acc[4], sh_red[0]);
}

__global__ void k_finalize(const float* __restrict__ acc, float* __restrict__ out) {
  if (threadIdx.x != 0 || blockIdx.x != 0) return;
  const float invT = 1.f / (float)T_STEPS;
  out[0] = acc[2] * invT - 0.01f * (acc[3] * invT) + 0.5f * (0.5f * acc[4] * invT);
}

extern "C" void kernel_launch(void* const* d_in, const int* in_sizes, int n_in,
                              void* d_out, int out_size, void* d_ws, size_t ws_size,
                              hipStream_t stream) {
  const float* obs      = (const float*)d_in[0];
  const int*   actions  = (const int*)  d_in[1];
  const float* logprobs = (const float*)d_in[2];
  const float* rewards  = (const float*)d_in[3];
  const float* terms    = (const float*)d_in[4];
  const float* values   = (const float*)d_in[5];
  const float* W1 = (const float*)d_in[6];
  const float* b1 = (const float*)d_in[7];
  const float* W2 = (const float*)d_in[8];
  const float* b2 = (const float*)d_in[9];
  const float* Wa = (const float*)d_in[10];
  const float* ba = (const float*)d_in[11];
  const float* Wc = (const float*)d_in[12];
  const float* bc = (const float*)d_in[13];

  float* ws  = (float*)d_ws;
  float* acc = ws;
  float* adv = ws + 8;
  float* P   = adv + T_STEPS;
  float* Q   = P + NCH;
  float* C   = Q + NCH;
  __bf16* W1s = (__bf16*)(C + NCH);      // 16B-aligned: 4*(8+65536+768) = 265248
  __bf16* W2s = W1s + W1S_ELEMS;
  __bf16* Whs = W2s + W2S_ELEMS;

  k_zero<<<1, 32, 0, stream>>>(acc);
  k_prep<<<(W1S_ELEMS + 255) / 256, 256, 0, stream>>>(W1, W2, Wa, Wc, W1s, W2s, Whs);
  k_gae_compose<<<(NCH + 127) / 128, 128, 0, stream>>>(rewards, terms, values, P, Q);
  k_gae_carry<<<1, 32, 0, stream>>>(P, Q, C);
  k_gae_apply<<<(NCH + 127) / 128, 128, 0, stream>>>(rewards, terms, values, C, adv, acc);
  k_gae_stats<<<1, 32, 0, stream>>>(acc);
  k_ppo_main<<<T_STEPS / 64, 128, 0, stream>>>(obs, actions, logprobs, values,
                                               W1s, b1, W2s, b2, Whs, ba, bc,
                                               adv, acc);
  k_finalize<<<1, 32, 0, stream>>>(acc, (float*)d_out);
}
